// MultiHeadAttention_54185307406568
// MI455X (gfx1250) — matrix operand order repaired and checked by reading, not compiled, not run on hardware
//
#include <hip/hip_runtime.h>
#include <hip/hip_bf16.h>

#define EMBED 1024
#define HEADS 16
#define HD    64
#define BATCH 2
#define SEQ   2048
#define MTOT  (BATCH * SEQ)   // 4096

#define LOG2E 1.44269504088896340736f

typedef __attribute__((ext_vector_type(16))) __bf16 v16bf;
typedef __attribute__((ext_vector_type(8)))  float  v8f;

// native f32->bf16 (RNE) -- backend emits v_cvt(_pk)_bf16_f32
static __device__ __forceinline__ __bf16 f2bf(float f) { return (__bf16)f; }

static __device__ __forceinline__ float fast_exp2(float x) {
#if __has_builtin(__builtin_amdgcn_exp2f)
  return __builtin_amdgcn_exp2f(x);   // bare v_exp_f32
#else
  return exp2f(x);
#endif
}

// ---- CDNA5 async global->LDS copy (GLOBAL_LOAD_ASYNC_TO_LDS_B128, ASYNCcnt) ----
static __device__ __forceinline__ unsigned lds_off(const void* p) {
  // generic LDS address: low 32 bits are the wave-relative LDS byte offset
  return (unsigned)(size_t)p;
}
static __device__ __forceinline__ void async_cp16(unsigned ldsoff, const void* g) {
  asm volatile("global_load_async_to_lds_b128 %0, %1, off"
               :: "v"(ldsoff), "v"(g) : "memory");
}
static __device__ __forceinline__ void wait_async0() {
#if __has_builtin(__builtin_amdgcn_s_wait_asynccnt)
  __builtin_amdgcn_s_wait_asynccnt(0);
#else
  asm volatile("s_wait_asynccnt 0" ::: "memory");
#endif
}

// ---- DPP16 row reductions (16-lane groups; rotate-allreduce) ----
template <int N>
static __device__ __forceinline__ float row_ror(float x) {
  return __int_as_float(__builtin_amdgcn_update_dpp(
      0, __float_as_int(x), 0x120 + N /*ROW_ROR:N*/, 0xF, 0xF, true));
}
static __device__ __forceinline__ float rowmax16(float v) {
  v = fmaxf(v, row_ror<8>(v));
  v = fmaxf(v, row_ror<4>(v));
  v = fmaxf(v, row_ror<2>(v));
  v = fmaxf(v, row_ror<1>(v));
  return v;
}
static __device__ __forceinline__ float rowsum16(float v) {
  v += row_ror<8>(v);
  v += row_ror<4>(v);
  v += row_ror<2>(v);
  v += row_ror<1>(v);
  return v;
}

// A-fragment (16x32 bf16, ISA 7.12.2): lane -> m = lane&15, khalf = lane>>4.
// elements 0..7  <- K = k0 + khalf*8 + (0..7); 8..15 <- K = k0 + 16 + khalf*8 + (0..7)
static __device__ __forceinline__ v16bf ld_fragA(const __bf16* base, int stride,
                                                 int row, int k0) {
  const int lane = threadIdx.x & 31;
  const __bf16* p = base + (size_t)(row + (lane & 15)) * stride + k0 + ((lane >> 4) << 3);
  union { uint4 u[2]; v16bf v; } x;
  x.u[0] = *(const uint4*)(p);
  x.u[1] = *(const uint4*)(p + 16);
  return x.v;
}
// B-fragment (32x16 bf16) from N-major LDS tile Bt[n][k]:
// lane -> n = lane&15, khalf = lane>>4; elements 0..7 <- K = k0 + khalf*8 + (0..7); 8..15 <- K = k0 + 16 + khalf*8 + (0..7)
static __device__ __forceinline__ v16bf ld_fragB(const __bf16* base, int stride,
                                                 int row, int k0) {
  const int lane = threadIdx.x & 31;
  const __bf16* p = base + (size_t)(row + (lane & 15)) * stride + k0 + ((lane >> 4) << 3);
  union { uint4 u[2]; v16bf v; } x;
  x.u[0] = *(const uint4*)(p);
  x.u[1] = *(const uint4*)(p + 16);
  return x.v;
}

// fp32 -> bf16 with optional pre-scale (used to fold 1/sqrt(E)*log2e into Wq)
__global__ __launch_bounds__(256) void cvt_f32_to_bf16(const float* __restrict__ src,
                                                       __bf16* __restrict__ dst,
                                                       int n, float scale) {
  int i = (blockIdx.x * 256 + threadIdx.x) * 4;
  if (i + 3 < n) {
    float4 f = *(const float4*)(src + i);
    dst[i + 0] = f2bf(f.x * scale);
    dst[i + 1] = f2bf(f.y * scale);
    dst[i + 2] = f2bf(f.z * scale);
    dst[i + 3] = f2bf(f.w * scale);
  }
}

// C = A(MxK,bf16) * Bw^T, Bw row-major [N][K]. 128x128 block tile, 8 wave32,
// ping-pong LDS: async DMA of tile i+1 overlaps WMMA on tile i.
// TRANS_OUT: scatter bf16 output into [b][h][d][s] (for V). OUT_F32: fp32 output.
template <bool TRANS_OUT, bool OUT_F32>
__global__ __launch_bounds__(256) void gemm128(const __bf16* __restrict__ A,
                                               const __bf16* __restrict__ Bw,
                                               void* __restrict__ Cp,
                                               int M, int N, int K) {
  constexpr int KP = 40;  // 32 + 8 pad (80B pitch, 16B aligned)
  __shared__ __bf16 sA[2][128 * KP];
  __shared__ __bf16 sB[2][128 * KP];

  const int tid  = threadIdx.x;
  const int lane = tid & 31;
  const int wave = tid >> 5;
  const int wr   = wave >> 2;   // 0..1
  const int wc   = wave & 3;    // 0..3
  const int bm   = blockIdx.y * 128;
  const int bn   = blockIdx.x * 128;

  const int srow = tid >> 1;         // 0..127
  const int sseg = (tid & 1) << 4;   // 0 or 16

  const __bf16* ga = A  + (size_t)(bm + srow) * K + sseg;
  const __bf16* gb = Bw + (size_t)(bn + srow) * K + sseg;
  unsigned la[2], lb[2];
  la[0] = lds_off(&sA[0][srow * KP + sseg]);
  la[1] = lds_off(&sA[1][srow * KP + sseg]);
  lb[0] = lds_off(&sB[0][srow * KP + sseg]);
  lb[1] = lds_off(&sB[1][srow * KP + sseg]);

  v8f acc[4][2];
  for (int rt = 0; rt < 4; rt++)
    for (int ct = 0; ct < 2; ct++)
      for (int i = 0; i < 8; i++) acc[rt][ct][i] = 0.0f;

  // prologue: stage tile 0
  async_cp16(la[0],      ga);
  async_cp16(la[0] + 16, ga + 8);
  async_cp16(lb[0],      gb);
  async_cp16(lb[0] + 16, gb + 8);
  wait_async0();
  __syncthreads();

  int buf = 0;
  for (int k0 = 0; k0 < K; k0 += 32, buf ^= 1) {
    if (k0 + 32 < K) {  // async-stage next tile into the other buffer
      const int nb = buf ^ 1;
      async_cp16(la[nb],      ga + k0 + 32);
      async_cp16(la[nb] + 16, ga + k0 + 40);
      async_cp16(lb[nb],      gb + k0 + 32);
      async_cp16(lb[nb] + 16, gb + k0 + 40);
      if (k0 + 64 < K) {  // prefetch tile after that -> global_prefetch_b8
        __builtin_prefetch(ga + k0 + 64, 0, 3);
        __builtin_prefetch(gb + k0 + 64, 0, 3);
      }
    }

    v16bf af[4], bfg[2];
#pragma unroll
    for (int rt = 0; rt < 4; rt++)
      af[rt] = ld_fragA(sA[buf], KP, (wr << 6) + (rt << 4), 0);
#pragma unroll
    for (int ct = 0; ct < 2; ct++)
      bfg[ct] = ld_fragB(sB[buf], KP, (wc << 5) + (ct << 4), 0);
#pragma unroll
    for (int rt = 0; rt < 4; rt++)
#pragma unroll
      for (int ct = 0; ct < 2; ct++)
        acc[rt][ct] = __builtin_amdgcn_wmma_f32_16x16x32_bf16(
            false, af[rt], false, bfg[ct], (short)0, acc[rt][ct], false, false);

    wait_async0();     // next tile's DMA landed
    __syncthreads();   // all waves done reading this tile
  }

  const int rbase = bm + (wr << 6);
  const int cbase = bn + (wc << 5);
  const int hi = (lane >> 4) << 3;  // lanes 16..31 hold rows M = 8 + r
  const int ln = lane & 15;
#pragma unroll
  for (int rt = 0; rt < 4; rt++)
#pragma unroll
    for (int ct = 0; ct < 2; ct++)
#pragma unroll
      for (int r = 0; r < 8; r++) {
        const int m = rbase + (rt << 4) + r + hi;
        const int n = cbase + (ct << 4) + ln;
        if constexpr (TRANS_OUT) {
          // (m = b*SEQ + s, n = h*HD + d) -> Vt[b][h][d][s]
          const int b = m >> 11, s = m & (SEQ - 1);
          const int h = n >> 6,  d = n & (HD - 1);
          ((__bf16*)Cp)[((size_t)(b * HEADS + h) * HD + d) * SEQ + s] =
              f2bf(acc[rt][ct][r]);
        } else if constexpr (OUT_F32) {
          ((float*)Cp)[(size_t)m * N + n] = acc[rt][ct][r];
        } else {
          ((__bf16*)Cp)[(size_t)m * N + n] = f2bf(acc[rt][ct][r]);
        }
      }
}

// Flash attention: one block per (64-row q tile, b, h). 128 threads = 4 waves;
// wave w owns the 16-row stripe [16w,16w+16) with its own online softmax.
// Q was projected with Wq pre-scaled by (1/sqrt(E))*log2e, so scores are
// already in the log2 domain; exp == bare v_exp_f32.
// Vt is pre-transposed [b][h][d][s]; K/V tiles ping-pong via async DMA.
__global__ __launch_bounds__(128) void attn64(const __bf16* __restrict__ Qb,
                                              const __bf16* __restrict__ Kb,
                                              const __bf16* __restrict__ Vt,
                                              __bf16* __restrict__ Ctx,
                                              const int* __restrict__ maskp) {
  constexpr int SP = 72;  // 64 + 8 pad
  __shared__ __bf16 sQ[64 * SP];
  __shared__ __bf16 sK[2][64 * SP];    // [key][d] -> B layout for Q*K^T
  __shared__ __bf16 sV[2][64 * SP];    // [d][key] -> B layout for P*V
  __shared__ __bf16 sP[4 * 16 * SP];   // per-wave P bounce buffer

  const int tid  = threadIdx.x;
  const int lane = tid & 31;
  const int wave = tid >> 5;      // 0..3
  const int qt   = blockIdx.x;    // q tile index
  const int bh   = blockIdx.y;
  const int b    = bh / HEADS;
  const int h    = bh % HEADS;

  const int srow = tid >> 1;        // 0..63
  const int sseg = (tid & 1) << 5;  // 0 or 32

  const __bf16* gq = Qb + (size_t)(b * SEQ + qt * 64 + srow) * EMBED + h * HD + sseg;
  const __bf16* gk = Kb + (size_t)(b * SEQ + srow) * EMBED + h * HD + sseg;
  const __bf16* gv = Vt + ((size_t)bh * HD + srow) * SEQ + sseg;
  const unsigned lq = lds_off(&sQ[srow * SP + sseg]);
  unsigned lk[2], lv[2];
  lk[0] = lds_off(&sK[0][srow * SP + sseg]);
  lk[1] = lds_off(&sK[1][srow * SP + sseg]);
  lv[0] = lds_off(&sV[0][srow * SP + sseg]);
  lv[1] = lds_off(&sV[1][srow * SP + sseg]);

  // prologue: stage Q and tile j=0 (all async)
#pragma unroll
  for (int i = 0; i < 4; i++) async_cp16(lq + 16 * i, gq + 8 * i);
#pragma unroll
  for (int i = 0; i < 4; i++) async_cp16(lk[0] + 16 * i, gk + 8 * i);
#pragma unroll
  for (int i = 0; i < 4; i++) async_cp16(lv[0] + 16 * i, gv + 8 * i);
  wait_async0();
  __syncthreads();

  v16bf aq[2];
  aq[0] = ld_fragA(sQ, SP, wave << 4, 0);
  aq[1] = ld_fragA(sQ, SP, wave << 4, 32);

  float m_run[8], l_run[8];
#pragma unroll
  for (int r = 0; r < 8; r++) { m_run[r] = -3.0e38f; l_run[r] = 0.0f; }
  v8f o[4];
  for (int dt = 0; dt < 4; dt++)
    for (int i = 0; i < 8; i++) o[dt][i] = 0.0f;

  const int domask = maskp[0];
  const int jmax = domask ? qt : (SEQ / 64 - 1);
  const int hi = (lane >> 4) << 3;
  const int ln = lane & 15;
  const float mask_add = -100000.0f * LOG2E;  // mask in log2 domain

  int buf = 0;
  for (int j = 0; j <= jmax; j++, buf ^= 1) {
    if (j < jmax) {  // async-stage next K/V tile into the other buffer
      const int nb = buf ^ 1;
      const __bf16* k = gk + (size_t)((j + 1) * 64) * EMBED;
      const __bf16* v = gv + (j + 1) * 64;
#pragma unroll
      for (int i = 0; i < 4; i++) async_cp16(lk[nb] + 16 * i, k + 8 * i);
#pragma unroll
      for (int i = 0; i < 4; i++) async_cp16(lv[nb] + 16 * i, v + 8 * i);
      if (j + 1 < jmax) {  // prefetch the tile after that
        __builtin_prefetch(k + (size_t)64 * EMBED, 0, 3);
        __builtin_prefetch(v + 64, 0, 3);
      }
    }

    // scores stripe: 16 x 64 (already scaled by 1/sqrt(E)*log2e via Wq)
    v8f sc[4];
    for (int ct = 0; ct < 4; ct++)
      for (int i = 0; i < 8; i++) sc[ct][i] = 0.0f;
#pragma unroll
    for (int ct = 0; ct < 4; ct++)
#pragma unroll
      for (int kk = 0; kk < 2; kk++) {
        v16bf bk = ld_fragB(sK[buf], SP, ct << 4, kk << 5);
        sc[ct] = __builtin_amdgcn_wmma_f32_16x16x32_bf16(
            false, aq[kk], false, bk, (short)0, sc[ct], false, false);
      }

    if (domask && j == qt) {  // diagonal tile: mask above diagonal
#pragma unroll
      for (int ct = 0; ct < 4; ct++) {
        const int col = (ct << 4) + ln;
#pragma unroll
        for (int r = 0; r < 8; r++) {
          const int row = (wave << 4) + r + hi;
          if (col > row) sc[ct][r] += mask_add;
        }
      }
    }

    // online softmax in log2 domain; row stats via DPP16 reductions
    float mnew[8], psc[8];
#pragma unroll
    for (int r = 0; r < 8; r++) {
      float v = fmaxf(fmaxf(sc[0][r], sc[1][r]), fmaxf(sc[2][r], sc[3][r]));
      v = rowmax16(v);
      mnew[r] = fmaxf(m_run[r], v);
      psc[r]  = fast_exp2(m_run[r] - mnew[r]);
    }
#pragma unroll
    for (int ct = 0; ct < 4; ct++)
#pragma unroll
      for (int r = 0; r < 8; r++) sc[ct][r] = fast_exp2(sc[ct][r] - mnew[r]);
#pragma unroll
    for (int r = 0; r < 8; r++) {
      float s = rowsum16(sc[0][r] + sc[1][r] + sc[2][r] + sc[3][r]);
      l_run[r] = l_run[r] * psc[r] + s;
      m_run[r] = mnew[r];
    }
#pragma unroll
    for (int dt = 0; dt < 4; dt++)
#pragma unroll
      for (int r = 0; r < 8; r++) o[dt][r] *= psc[r];

    // bounce P through LDS: C layout -> A fragment layout, bf16
    __bf16* pw = &sP[wave * 16 * SP];
#pragma unroll
    for (int ct = 0; ct < 4; ct++)
#pragma unroll
      for (int r = 0; r < 8; r++)
        pw[(r + hi) * SP + (ct << 4) + ln] = f2bf(sc[ct][r]);
    __syncthreads();

    // O += P * V
#pragma unroll
    for (int kk = 0; kk < 2; kk++) {
      v16bf ap = ld_fragA(pw, SP, 0, kk << 5);
#pragma unroll
      for (int dt = 0; dt < 4; dt++) {
        v16bf bv = ld_fragB(sV[buf], SP, dt << 4, kk << 5);
        o[dt] = __builtin_amdgcn_wmma_f32_16x16x32_bf16(
            false, ap, false, bv, (short)0, o[dt], false, false);
      }
    }

    wait_async0();     // next tile's DMA landed
    __syncthreads();   // all waves done with this tile (and sP reads done)
  }

  // epilogue: normalize and write context (bf16) in (B,S,E) layout
#pragma unroll
  for (int dt = 0; dt < 4; dt++)
#pragma unroll
    for (int r = 0; r < 8; r++) {
      const int row = qt * 64 + (wave << 4) + r + hi;
      const int col = h * HD + (dt << 4) + ln;
      Ctx[(size_t)(b * SEQ + row) * EMBED + col] = f2bf(o[dt][r] / l_run[r]);
    }
}

extern "C" void kernel_launch(void* const* d_in, const int* in_sizes, int n_in,
                              void* d_out, int out_size, void* d_ws, size_t ws_size,
                              hipStream_t stream) {
  const float* key   = (const float*)d_in[0];
  const float* query = (const float*)d_in[1];
  const float* value = (const float*)d_in[2];
  const float* Wk    = (const float*)d_in[3];
  const float* Wq    = (const float*)d_in[4];
  const float* Wv    = (const float*)d_in[5];
  const float* Wo    = (const float*)d_in[6];
  const int*   mask  = (const int*)d_in[7];

  char* ws = (char*)d_ws;
  const size_t WSZ = (size_t)EMBED * EMBED * sizeof(__bf16);  // 2 MiB
  const size_t XSZ = (size_t)MTOT * EMBED * sizeof(__bf16);   // 8 MiB
  __bf16* wkb = (__bf16*)(ws);
  __bf16* wqb = (__bf16*)(ws + WSZ);
  __bf16* wvb = (__bf16*)(ws + 2 * WSZ);
  __bf16* wob = (__bf16*)(ws + 3 * WSZ);
  __bf16* xq  = (__bf16*)(ws + 4 * WSZ);             // bf16 activations
  __bf16* xk  = (__bf16*)(ws + 4 * WSZ + XSZ);
  __bf16* xv  = (__bf16*)(ws + 4 * WSZ + 2 * XSZ);
  __bf16* qb  = (__bf16*)(ws + 4 * WSZ + 3 * XSZ);   // projections
  __bf16* kb  = (__bf16*)(ws + 4 * WSZ + 4 * XSZ);
  __bf16* vtb = (__bf16*)(ws + 4 * WSZ + 5 * XSZ);   // V transposed [b][h][d][s]
  __bf16* ctx = (__bf16*)(ws + 4 * WSZ + 6 * XSZ);   // total 64 MiB

  const int nW = EMBED * EMBED, nX = MTOT * EMBED;
  const float qscale = (1.0f / 32.0f) * LOG2E;  // 1/sqrt(EMBED) * log2(e)
  dim3 cvb(256);
  cvt_f32_to_bf16<<<dim3(nW / 1024), cvb, 0, stream>>>(Wk, wkb, nW, 1.0f);
  cvt_f32_to_bf16<<<dim3(nW / 1024), cvb, 0, stream>>>(Wq, wqb, nW, qscale);
  cvt_f32_to_bf16<<<dim3(nW / 1024), cvb, 0, stream>>>(Wv, wvb, nW, 1.0f);
  cvt_f32_to_bf16<<<dim3(nW / 1024), cvb, 0, stream>>>(Wo, wob, nW, 1.0f);
  cvt_f32_to_bf16<<<dim3(nX / 1024), cvb, 0, stream>>>(query, xq, nX, 1.0f);
  cvt_f32_to_bf16<<<dim3(nX / 1024), cvb, 0, stream>>>(key,   xk, nX, 1.0f);
  cvt_f32_to_bf16<<<dim3(nX / 1024), cvb, 0, stream>>>(value, xv, nX, 1.0f);

  dim3 gb(256);
  dim3 gg(EMBED / 128, MTOT / 128);
  gemm128<false, false><<<gg, gb, 0, stream>>>(xq, wqb, qb,   MTOT, EMBED, EMBED);
  gemm128<false, false><<<gg, gb, 0, stream>>>(xk, wkb, kb,   MTOT, EMBED, EMBED);
  gemm128<true,  false><<<gg, gb, 0, stream>>>(xv, wvb, vtb,  MTOT, EMBED, EMBED);

  dim3 ab(128);
  dim3 ag(SEQ / 64, BATCH * HEADS);
  attn64<<<ag, ab, 0, stream>>>(qb, kb, vtb, ctx, mask);

  gemm128<false, true><<<gg, gb, 0, stream>>>(ctx, wob, d_out, MTOT, EMBED, EMBED);
}
